// SnoringClassificationLayer_52166672778111
// MI455X (gfx1250) — matrix operand order repaired and checked by reading, not compiled, not run on hardware
//
#include <hip/hip_runtime.h>

// CDNA5 / gfx1250, wave32. Single-wave kernel: workload is ~7KB of input and
// 2 output scalars -> pure latency problem. The sliding-window reductions are
// done with V_WMMA_F32_16X16X32_F16 as a banded-Toeplitz mat-vec:
//   D[m][n] = sum_k A[m][k] * B[k][n],  A = width-5 band (0/1),
//   B[k][n] = x[k + 16n]  (n = 0..2)  ==>  D[m][n] = window_sum(x)[m + 16n]
// Operands are {0,1} and integer-scaled scores -> exact in f16 * f16 -> f32 acc.
// Final count/top-2 reduction is fully branchless + a 5-step shfl_xor butterfly.

typedef __attribute__((ext_vector_type(16))) _Float16 v16h;
typedef __attribute__((ext_vector_type(8)))  float    v8f;

#define NF         42
#define WIN        5            // GROUP_SIZE
#define MIN_VALID  3
#define REQ_TRUE   2
#define NGROUPS    (NF - WIN + 1)   // 38
#define NEGINF     (-3.0e38f)

__global__ __launch_bounds__(32)
void snoring_cls_kernel(const float* __restrict__ res,    // [42,42]
                        const float* __restrict__ comb,   // [3,3,3] (cls, lo, hi)
                        const float* __restrict__ score,  // [3]
                        float* __restrict__ out)          // [2] = (final_j, final_s)
{
    __shared__ float lds_fj[64];   // frame_j as 0/1
    __shared__ float lds_fs[64];   // frame_s scaled by 100 (integer-valued)
    __shared__ float lds_cnt[48];  // window counts   (padded: g up to 47 written)
    __shared__ float lds_sum[48];  // window sums x100 (padded)

    const int lane = threadIdx.x;  // 0..31, one wave

    // ---- Stage 1: per-frame combination check (2 frames per lane) ----------
    for (int rep = 0; rep < 2; ++rep) {
        const int f = lane + 32 * rep;       // 0..63
        float fj = 0.0f, fs = 0.0f;
        if (f < NF) {
            bool any = false;
            int  first = 0;
            for (int j = 0; j < 3; ++j) {
                bool ok = true;
                for (int k = 0; k < 3; ++k) {
                    const float* c3 = comb + (j * 3 + k) * 3;
                    const int   cls = (int)c3[0];
                    const float v   = res[f * NF + cls];
                    ok = ok && (v >= c3[1]) && (v <= c3[2]);
                }
                if (ok && !any) { any = true; first = j; }
            }
            if (any) {
                fj = 1.0f;
                // jnp.round is half-to-even -> rintf (default RNE mode).
                fs = rintf(score[first] * 100.0f);   // keep x100 scale: exact in f16
            }
        }
        lds_fj[f] = fj;   // f in 0..63, zero-padded for f >= 42
        lds_fs[f] = fs;
    }
    __syncthreads();

    // ---- Stage 2: width-5 sliding sums via WMMA ----------------------------
    // A-matrix (16x32 f16): lanes 0-15 -> M=lane, element e -> K = e,
    // lanes 16-31 -> K = 16 + e (same contiguous K order as B).
    // B-matrix (32x16 f16): lane = column n (lanes 16-31 -> K+16), element e -> K.
    const int m   = lane & 15;
    const int n   = lane & 15;
    const int grp = lane >> 4;

    v16h a, bj, bs;
    #pragma unroll
    for (int e = 0; e < 16; ++e) {
        const int v = e >> 1, h = e & 1;
        const int ka = 16 * grp + e;
        a[e] = (_Float16)((ka >= m && ka <= m + WIN - 1) ? 1.0f : 0.0f);

        const int kb = 16 * grp + e;          // B row index
        const int xi = kb + 16 * n;           // <= 31 + 32 = 63 when n < 3
        const float vj = (n < 3) ? lds_fj[xi] : 0.0f;
        const float vs = (n < 3) ? lds_fs[xi] : 0.0f;
        bj[e] = (_Float16)vj;
        bs[e] = (_Float16)vs;
    }

    v8f c0 = {}, c1 = {};
    // (neg_a, A, neg_b, B, c_mod, C, reuse_a, reuse_b)
    v8f dcnt = __builtin_amdgcn_wmma_f32_16x16x32_f16(false, a, false, bj,
                                                      (short)0, c0, false, false);
    v8f dsum = __builtin_amdgcn_wmma_f32_16x16x32_f16(false, a, false, bs,
                                                      (short)0, c1, false, false);

    // Scatter D[m][n] = y[m + 16n] (C/D layout: VGPR r -> M = r + 8*grp, N = lane%16).
    // Max g for n<3 is 7 + 8 + 32 = 47 < 48 -> store unconditionally into the
    // padded arrays (no per-store exec toggles); only g < 38 is read back.
    if (n < 3) {
        const int gbase = 8 * grp + 16 * n;
        #pragma unroll
        for (int r = 0; r < 8; ++r) {
            lds_cnt[gbase + r] = dcnt[r];
            lds_sum[gbase + r] = dsum[r];
        }
    }
    __syncthreads();

    // ---- Stage 3: branchless count + top-2 reduction -----------------------
    // Each lane evaluates 2 group candidates with unconditional (clamped) LDS
    // loads and cndmask selection, then a 5-step xor-butterfly merges
    // (true_count, top1, top2) across the wave.
    int   tc = 0;
    float cand0, cand1;
    {
        const int g0 = lane;                       // 0..31, always < NGROUPS
        const int g1 = lane + 32;                  // 32..63, valid only < 38
        const int g1c = (g1 < NGROUPS) ? g1 : 0;   // clamp for safe LDS read
        const float c0v = lds_cnt[g0];
        const float s0v = lds_sum[g0];
        const float c1v = lds_cnt[g1c];
        const float s1v = lds_sum[g1c];
        const bool v0 = (c0v >= (float)MIN_VALID - 0.5f);
        const bool v1 = (g1 < NGROUPS) && (c1v >= (float)MIN_VALID - 0.5f);
        cand0 = v0 ? s0v * 0.01f : NEGINF;         // undo x100 scale
        cand1 = v1 ? s1v * 0.01f : NEGINF;
        tc = (v0 ? 1 : 0) + (v1 ? 1 : 0);
    }
    float m1 = fmaxf(cand0, cand1);
    float m2 = fminf(cand0, cand1);

    #pragma unroll
    for (int off = 16; off >= 1; off >>= 1) {
        const int   oc = __shfl_xor(tc, off, 32);
        const float o1 = __shfl_xor(m1, off, 32);
        const float o2 = __shfl_xor(m2, off, 32);
        tc += oc;
        const float hi = fmaxf(m1, o1);
        const float lo = fminf(m1, o1);
        m2 = fmaxf(lo, fmaxf(m2, o2));
        m1 = hi;
    }

    if (lane == 0) {
        const bool fj = (tc >= REQ_TRUE);
        out[0] = fj ? 1.0f : 0.0f;
        out[1] = fj ? (m1 + m2) : 0.0f;
    }
}

extern "C" void kernel_launch(void* const* d_in, const int* in_sizes, int n_in,
                              void* d_out, int out_size, void* d_ws, size_t ws_size,
                              hipStream_t stream) {
    (void)in_sizes; (void)n_in; (void)out_size; (void)d_ws; (void)ws_size;
    const float* res   = (const float*)d_in[0];   // snoring_result  [42*42] f32
    const float* comb  = (const float*)d_in[1];   // combination_list [27]   f32
    const float* score = (const float*)d_in[2];   // snoring_score    [3]    f32
    float* out = (float*)d_out;                   // [final_j, final_s] f32

    snoring_cls_kernel<<<1, 32, 0, stream>>>(res, comb, score, out);
}
